// Block_ATC_21689584844859
// MI455X (gfx1250) — matrix operand order repaired and checked by reading, not compiled, not run on hardware
//
#include <hip/hip_runtime.h>
#include <hip/hip_bf16.h>

#define BB 64
#define TT 512
#define CC 256
#define INF_F 1e9f
#define EPS_F 1e-8f

typedef __attribute__((ext_vector_type(16))) _Float16 v16h;
typedef __attribute__((ext_vector_type(8)))  _Float16 v8h;
typedef __attribute__((ext_vector_type(8)))  float    v8f;

// ---------------- Kernel 1: row-normalize x -> f16 Xn ----------------
// One 256-thread block per row (B*T rows). Each thread owns one channel.
__global__ void normalize_rows(const float* __restrict__ x,
                               _Float16* __restrict__ xn) {
  const int row = blockIdx.x;      // 0 .. B*T-1
  const int t   = threadIdx.x;     // 0 .. CC-1
  __shared__ float red[CC];
  float v = x[(size_t)row * CC + t];
  red[t] = v * v;
  __syncthreads();
  for (int s = CC / 2; s > 0; s >>= 1) {
    if (t < s) red[t] += red[t + s];
    __syncthreads();
  }
  float inv = 1.0f / (sqrtf(red[0]) + EPS_F);
  xn[(size_t)row * CC + t] = (_Float16)(v * inv);
}

// ---------------- Kernel 2: D = 1 - Xn * Xn^T via WMMA f16 ----------------
// One wave32 per 16x16 output tile. K loop: 256 / 32 = 8 WMMA ops.
__global__ void wmma_cosine_dist(const _Float16* __restrict__ xn,
                                 float* __restrict__ D) {
  const int j0   = blockIdx.x * 16;
  const int i0   = blockIdx.y * 16;
  const int b    = blockIdx.z;
  const int lane = threadIdx.x;                 // 0..31, wave32
  const _Float16* Xb = xn + (size_t)b * TT * CC;

  const int m     = lane & 15;                  // row (A) / col (B) index
  const int hi    = (lane >> 4) & 1;            // lane group
  const int kselA = hi ? 16 : 0;                // A: K sub-block base
  const int koffB = hi ? 16 : 0;                // B: K half base

  const _Float16* arow = Xb + (size_t)(i0 + m) * CC;  // A tile rows i0..i0+15
  const _Float16* brow = Xb + (size_t)(j0 + m) * CC;  // B cols = Xn rows j0..j0+15

  v8f acc = {};
#pragma unroll
  for (int k0 = 0; k0 < CC; k0 += 32) {
    // A fragment: halves [0..7] = K(k0+kselA..+7), [8..15] = K(k0+kselA+8..+15)
    v8h alo = *(const v8h*)(arow + k0 + kselA);
    v8h ahi = *(const v8h*)(arow + k0 + kselA + 8);
    v16h a;
#pragma unroll
    for (int q = 0; q < 8; ++q) { a[q] = alo[q]; a[q + 8] = ahi[q]; }
    // B fragment: 16 contiguous K values of column (j0+m), half picked by lane group
    v16h bfrag = *(const v16h*)(brow + k0 + koffB);
    acc = __builtin_amdgcn_wmma_f32_16x16x32_f16(
        false, a, false, bfrag, (short)0, acc, false, false);
  }

  float* Db = D + (size_t)b * TT * TT;
  const int mbase = hi ? 8 : 0;
#pragma unroll
  for (int r = 0; r < 8; ++r) {
    int row = i0 + mbase + r;
    int col = j0 + m;
    float dv = 1.0f - acc[r];
    if (row == col) dv = INF_F;                 // diagonal -> INF
    Db[(size_t)row * TT + col] = dv;
  }
}

// ---------------- Kernel 3: average-linkage agglomerative merge ----------------
// One 512-thread block (16 wave32s) per batch. Maintains per-row minima in LDS
// so each merge's argmin is a 512-wide tree reduce instead of a 256K scan.
__global__ void agglom_cluster(float* __restrict__ D,
                               const int* __restrict__ ncl,
                               float* __restrict__ out) {
  const int b = blockIdx.x;
  const int t = threadIdx.x;                    // 0..TT-1
  float* Db = D + (size_t)b * TT * TT;

  __shared__ float rowmin[TT];
  __shared__ int   rowarg[TT];
  __shared__ float sizes[TT];
  __shared__ int   assign[TT];
  __shared__ float redv[TT];
  __shared__ int   redi[TT];
  __shared__ int   mi, mj;

  int k = ncl[0];
  if (k > TT) k = TT;
  if (k < 1) k = 1;
  const int n_merges = TT - k;

  sizes[t]  = 1.0f;
  assign[t] = t;
  { // initial per-row minima (first-occurrence argmin: strict <, ascending c)
    float mv = INF_F; int ma = 0;
    const float* dr = Db + (size_t)t * TT;
    for (int c = 0; c < TT; ++c) {
      float v = dr[c];
      if (v < mv) { mv = v; ma = c; }
    }
    rowmin[t] = mv; rowarg[t] = ma;
  }
  __syncthreads();

  for (int it = 0; it < n_merges; ++it) {
    // ---- global argmin over row minima, tie-break = smallest flat index ----
    redv[t] = rowmin[t]; redi[t] = t;
    __syncthreads();
    for (int s = TT / 2; s > 0; s >>= 1) {
      if (t < s) {
        float va = redv[t], vb = redv[t + s];
        int   ra = redi[t], rb = redi[t + s];
        bool take = (vb < va) ||
                    (vb == va &&
                     (rb * TT + rowarg[rb]) < (ra * TT + rowarg[ra]));
        if (take) { redv[t] = vb; redi[t] = rb; }
      }
      __syncthreads();
    }
    if (t == 0) { mi = redi[0]; mj = rowarg[redi[0]]; }  // flat tie-break => mi < mj
    __syncthreads();
    const int i = mi, j = mj;
    const float ni = sizes[i], nj = sizes[j];

    // ---- size-weighted merge of rows/cols i,j: read phase, then write phase ----
    float di = Db[(size_t)i * TT + t];
    float dj = Db[(size_t)j * TT + t];
    __syncthreads();                            // all reads done before any write
    float newv = (ni * di + nj * dj) / (ni + nj);
    float fin  = (t == i || t == j) ? INF_F : newv;
    Db[(size_t)i * TT + t] = fin;
    Db[(size_t)t * TT + i] = fin;
    Db[(size_t)j * TT + t] = INF_F;
    Db[(size_t)t * TT + j] = INF_F;
    if (t == 0) { sizes[i] = ni + nj; sizes[j] = 0.0f; }
    if (assign[t] == j) assign[t] = i;
    __syncthreads();

    // ---- repair cached row minima ----
    if (t == j) {
      rowmin[t] = INF_F; rowarg[t] = 0;         // row j is all-INF now
    } else if (t == i || rowarg[t] == i || rowarg[t] == j) {
      float mv = INF_F; int ma = 0;             // stale cache -> full rescan
      const float* dr = Db + (size_t)t * TT;
      for (int c = 0; c < TT; ++c) {
        float v = dr[c];
        if (v < mv) { mv = v; ma = c; }
      }
      rowmin[t] = mv; rowarg[t] = ma;
    } else {
      float nv = Db[(size_t)t * TT + i];        // only column i changed
      if (nv < rowmin[t] || (nv == rowmin[t] && i < rowarg[t])) {
        rowmin[t] = nv; rowarg[t] = i;
      }
    }
    __syncthreads();
  }

  // ---- canonical relabel: rep is always cluster's min index, so
  //      label[t] = #{ u < assign[t] : assign[u] == u } ----
  const int a = assign[t];
  int lbl = 0;
  for (int u = 0; u < a; ++u) lbl += (assign[u] == u) ? 1 : 0;
  out[(size_t)b * TT + t] = (float)lbl;
}

extern "C" void kernel_launch(void* const* d_in, const int* in_sizes, int n_in,
                              void* d_out, int out_size, void* d_ws, size_t ws_size,
                              hipStream_t stream) {
  (void)in_sizes; (void)n_in; (void)out_size; (void)ws_size;
  const float* x   = (const float*)d_in[0];
  const int*   ncl = (const int*)d_in[1];

  _Float16* xn = (_Float16*)d_ws;                                   // 16 MB
  float*    D  = (float*)((char*)d_ws +
                          (size_t)BB * TT * CC * sizeof(_Float16)); // 64 MB
  float*    out = (float*)d_out;

  normalize_rows  <<<dim3(BB * TT),               dim3(CC), 0, stream>>>(x, xn);
  wmma_cosine_dist<<<dim3(TT / 16, TT / 16, BB),  dim3(32), 0, stream>>>(xn, D);
  agglom_cluster  <<<dim3(BB),                    dim3(TT), 0, stream>>>(D, ncl, out);
}
